// GCNNet_52089363366228
// MI455X (gfx1250) — matrix operand order repaired and checked by reading, not compiled, not run on hardware
//
#include <hip/hip_runtime.h>
#include <hip/hip_bf16.h>

// ---------------------------------------------------------------------------
// GCN 2-layer forward for MI455X (gfx1250, wave32, WMMA)
//   x:[N,512] @ W1:[512,256] -> WMMA bf16 GEMM (f32 accum), 16x128 per wave
//   scatter-add aggregation with sym-norm (atomics, L2-resident gather)
//   relu -> [N,256] @ W2:[256,2] (VALU) -> aggregation -> d_out
// ---------------------------------------------------------------------------

typedef __attribute__((ext_vector_type(16))) __bf16 v16bf;
typedef __attribute__((ext_vector_type(8)))  float  v8f;

__device__ __forceinline__ unsigned short f2bf(float f) {
  unsigned int u = __float_as_uint(f);
  unsigned int r = u + 0x7FFFu + ((u >> 16) & 1u);   // round-to-nearest-even
  return (unsigned short)(r >> 16);
}

// ---- W1 -> bf16, pre-swizzled into WMMA B-fragment order --------------------
// Fragment (kt,nt): 32 lanes x 16 bf16. lane L: col = nt*16 + (L&15),
// elems j=0..15: K = kt*32 + (L>>4)*16 + j.  frag stride = 512 ushorts.
__global__ __launch_bounds__(256) void k_convW1(const float* __restrict__ W1,
                                                unsigned short* __restrict__ w1bf) {
  int t = blockIdx.x * 256 + threadIdx.x;          // 131072 total
  if (t >= 512 * 256) return;
  int frag = t >> 9;                               // 0..255
  int within = t & 511;
  int lane = within >> 4;
  int j = within & 15;
  int kt = frag >> 4;
  int nt = frag & 15;
  int col = nt * 16 + (lane & 15);
  int k = kt * 32 + ((lane >> 4) << 4) + j;
  w1bf[t] = f2bf(W1[k * 256 + col]);
}

// ---- degree / norm ----------------------------------------------------------
__global__ __launch_bounds__(256) void k_deg_init(float* __restrict__ deg, int N) {
  int i = blockIdx.x * 256 + threadIdx.x;
  if (i < N) deg[i] = 1.0f;                        // self-loop
}
__global__ __launch_bounds__(256) void k_deg_acc(const int* __restrict__ dst,
                                                 float* __restrict__ deg, int E) {
  int e = blockIdx.x * 256 + threadIdx.x;
  if (e < E) atomicAdd(&deg[dst[e]], 1.0f);
}
__global__ __launch_bounds__(256) void k_dinv(float* __restrict__ deg, int N) {
  int i = blockIdx.x * 256 + threadIdx.x;
  if (i < N) { float d = deg[i]; deg[i] = (d > 0.0f) ? rsqrtf(d) : 0.0f; }
}

// ---- GEMM1: h1 = x @ W1  (bf16 WMMA, f32 accum) -----------------------------
// One wave computes a 16x128 strip of h1 (8 accumulators); 2 waves per M-tile.
__global__ __launch_bounds__(256) void k_gemm1_wmma(const float* __restrict__ x,
                                                    const unsigned short* __restrict__ w1bf,
                                                    float* __restrict__ h1, int M) {
  const int lane = threadIdx.x & 31;
  const int gw = blockIdx.x * 8 + (threadIdx.x >> 5);
  const int mtiles = (M + 15) >> 4;
  if (gw >= mtiles * 2) return;
  const int mtile = gw >> 1;
  const int nstrip = gw & 1;                       // 0 or 1 -> cols [0,128) / [128,256)
  const int m0 = mtile * 16;
  const int n0 = nstrip * 128;

  int row = m0 + (lane & 15);
  if (row >= M) row = M - 1;                       // clamp loads (EXEC stays full)
  const int kb = (lane >> 4) << 4;                 // 0 or 16
  const float* xr = x + (size_t)row * 512;

  const v16bf* bp = (const v16bf*)w1bf;            // 32 v16bf per fragment
  const int ntb = nstrip * 8;

  v8f acc[8];
#pragma unroll
  for (int i = 0; i < 8; ++i) acc[i] = (v8f){};

  for (int kt = 0; kt < 16; ++kt) {
    const int k0 = kt * 32;
    const float* p0 = xr + k0 + kb;
    // prefetch next A tile (emits global_prefetch_b8)
    __builtin_prefetch(p0 + 32, 0, 0);

    float4 a0 = *(const float4*)(p0);
    float4 a1 = *(const float4*)(p0 + 4);
    float4 a2 = *(const float4*)(p0 + 8);
    float4 a3 = *(const float4*)(p0 + 12);

    union { v16bf v; unsigned short u[16]; } A;
    A.u[0]  = f2bf(a0.x); A.u[1]  = f2bf(a0.y); A.u[2]  = f2bf(a0.z); A.u[3]  = f2bf(a0.w);
    A.u[4]  = f2bf(a1.x); A.u[5]  = f2bf(a1.y); A.u[6]  = f2bf(a1.z); A.u[7]  = f2bf(a1.w);
    A.u[8]  = f2bf(a2.x); A.u[9]  = f2bf(a2.y); A.u[10] = f2bf(a2.z); A.u[11] = f2bf(a2.w);
    A.u[12] = f2bf(a3.x); A.u[13] = f2bf(a3.y); A.u[14] = f2bf(a3.z); A.u[15] = f2bf(a3.w);

    const v16bf* bk = bp + (size_t)(kt * 16 + ntb) * 32 + lane;
#pragma unroll
    for (int i = 0; i < 8; ++i) {
      v16bf b = bk[i * 32];
      acc[i] = __builtin_amdgcn_wmma_f32_16x16x32_bf16(false, A.v, false, b,
                                                       (short)0, acc[i], false, false);
    }
  }

  // D layout: VGPR r -> row m0+r (lanes 0-15) / m0+8+r (lanes 16-31); col = n0 + lane%16
  const int rbase = m0 + ((lane >> 4) << 3);
  const int col = n0 + (lane & 15);
  if (m0 + 16 <= M) {
    // full tile: branch-free store clause
    float* p = h1 + (size_t)rbase * 256 + col;
#pragma unroll
    for (int r = 0; r < 8; ++r) {
      float* pr = p + (size_t)r * 256;
#pragma unroll
      for (int i = 0; i < 8; ++i) pr[i * 16] = acc[i][r];
    }
  } else {
    // ragged final tile (not taken for M % 16 == 0)
#pragma unroll
    for (int r = 0; r < 8; ++r) {
      int rr = rbase + r;
      if (rr < M) {
        float* pr = h1 + (size_t)rr * 256 + col;
#pragma unroll
        for (int i = 0; i < 8; ++i) pr[i * 16] = acc[i][r];
      }
    }
  }
}

// ---- init aggregation dest with bias ---------------------------------------
__global__ __launch_bounds__(256) void k_init_b1(float* __restrict__ out,
                                                 const float* __restrict__ b1, long total) {
  long i = (long)blockIdx.x * 256 + threadIdx.x;
  if (i < total) out[i] = b1[i & 255];
}
__global__ __launch_bounds__(256) void k_init_b2(float* __restrict__ out,
                                                 const float* __restrict__ b2, int total) {
  int i = blockIdx.x * 256 + threadIdx.x;
  if (i < total) out[i] = b2[i & 1];
}

// ---- aggregation layer 1: one wave per edge, 256 channels -------------------
__global__ __launch_bounds__(256) void k_agg1(const float* __restrict__ h1,
                                              const int* __restrict__ src,
                                              const int* __restrict__ dst,
                                              const float* __restrict__ dinv,
                                              float* __restrict__ out, int E, int N) {
  const int lane = threadIdx.x & 31;
  const int e = blockIdx.x * 8 + (threadIdx.x >> 5);
  if (e >= E + N) return;
  int s, d; float nrm;
  if (e < E) { s = src[e]; d = dst[e]; nrm = dinv[s] * dinv[d]; }
  else       { s = d = e - E; float di = dinv[s]; nrm = di * di; }
  const float4* hs = (const float4*)(h1 + (size_t)s * 256);
  float* od = out + (size_t)d * 256 + lane * 8;
  float4 v0 = hs[lane * 2];
  float4 v1 = hs[lane * 2 + 1];
  atomicAdd(&od[0], v0.x * nrm); atomicAdd(&od[1], v0.y * nrm);
  atomicAdd(&od[2], v0.z * nrm); atomicAdd(&od[3], v0.w * nrm);
  atomicAdd(&od[4], v1.x * nrm); atomicAdd(&od[5], v1.y * nrm);
  atomicAdd(&od[6], v1.z * nrm); atomicAdd(&od[7], v1.w * nrm);
}

// ---- GEMM2 with fused relu: h2 = relu(hagg) @ W2  (256 -> 2) ----------------
__global__ __launch_bounds__(256) void k_gemm2(const float* __restrict__ hagg,
                                               const float* __restrict__ W2,
                                               float* __restrict__ h2, int N) {
  int n = blockIdx.x * 256 + threadIdx.x;
  if (n >= N) return;
  const float4* hr = (const float4*)(hagg + (size_t)n * 256);
  const float4* wp = (const float4*)W2;            // [256,2] -> 128 float4
  float a0 = 0.f, a1 = 0.f;
#pragma unroll 4
  for (int c = 0; c < 64; ++c) {
    float4 v = hr[c];
    float4 w01 = wp[2 * c];
    float4 w23 = wp[2 * c + 1];
    float v0 = fmaxf(v.x, 0.f), v1 = fmaxf(v.y, 0.f);
    float v2 = fmaxf(v.z, 0.f), v3 = fmaxf(v.w, 0.f);
    a0 += v0 * w01.x + v1 * w01.z + v2 * w23.x + v3 * w23.z;
    a1 += v0 * w01.y + v1 * w01.w + v2 * w23.y + v3 * w23.w;
  }
  h2[2 * n] = a0;
  h2[2 * n + 1] = a1;
}

// ---- aggregation layer 2: one thread per edge, 2 channels -------------------
__global__ __launch_bounds__(256) void k_agg2(const float* __restrict__ h2,
                                              const int* __restrict__ src,
                                              const int* __restrict__ dst,
                                              const float* __restrict__ dinv,
                                              float* __restrict__ out, int E, int N) {
  int e = blockIdx.x * 256 + threadIdx.x;
  if (e >= E + N) return;
  int s, d; float nrm;
  if (e < E) { s = src[e]; d = dst[e]; nrm = dinv[s] * dinv[d]; }
  else       { s = d = e - E; float di = dinv[s]; nrm = di * di; }
  float2 v = *(const float2*)(h2 + 2 * (size_t)s);
  atomicAdd(&out[2 * d],     v.x * nrm);
  atomicAdd(&out[2 * d + 1], v.y * nrm);
}

// ---------------------------------------------------------------------------
extern "C" void kernel_launch(void* const* d_in, const int* in_sizes, int n_in,
                              void* d_out, int out_size, void* d_ws, size_t ws_size,
                              hipStream_t stream) {
  const float* x  = (const float*)d_in[0];
  const int*   ei = (const int*)d_in[1];
  const float* W1 = (const float*)d_in[2];
  const float* b1 = (const float*)d_in[3];
  const float* W2 = (const float*)d_in[4];
  const float* b2 = (const float*)d_in[5];
  float* out = (float*)d_out;

  const int N = in_sizes[0] / 512;                 // 50000
  const int E = in_sizes[1] / 2;                   // 800000
  const int* src = ei;
  const int* dst = ei + E;

  // workspace carve-up (bytes, 1KB aligned)
  char* ws = (char*)d_ws;
  float*          dinv = (float*)(ws);                                  // N f32
  size_t off = ((size_t)N * 4 + 1023) & ~(size_t)1023;
  unsigned short* w1bf = (unsigned short*)(ws + off);                   // 512*256 bf16
  off += ((size_t)512 * 256 * 2 + 1023) & ~(size_t)1023;
  float*          h1   = (float*)(ws + off);                            // N*256 f32
  off += ((size_t)N * 256 * 4 + 1023) & ~(size_t)1023;
  float*          hagg = (float*)(ws + off);                            // N*256 f32
  off += ((size_t)N * 256 * 4 + 1023) & ~(size_t)1023;
  float*          h2   = (float*)(ws + off);                            // N*2 f32

  const int T = 256;
  // 1) W1 -> bf16 fragments
  k_convW1<<<(512 * 256 + T - 1) / T, T, 0, stream>>>(W1, w1bf);
  // 2) degrees -> dinv
  k_deg_init<<<(N + T - 1) / T, T, 0, stream>>>(dinv, N);
  k_deg_acc<<<(E + T - 1) / T, T, 0, stream>>>(dst, dinv, E);
  k_dinv<<<(N + T - 1) / T, T, 0, stream>>>(dinv, N);
  // 3) h1 = x @ W1 (WMMA, 16x128 strip per wave)
  {
    int waves = ((N + 15) / 16) * 2;
    k_gemm1_wmma<<<(waves + 7) / 8, T, 0, stream>>>(x, w1bf, h1, N);
  }
  // 4) hagg = b1; scatter-add messages
  {
    long tot = (long)N * 256;
    k_init_b1<<<(int)((tot + T - 1) / T), T, 0, stream>>>(hagg, b1, tot);
    k_agg1<<<(E + N + 7) / 8, T, 0, stream>>>(h1, src, dst, dinv, hagg, E, N);
  }
  // 5) h2 = relu(hagg) @ W2
  k_gemm2<<<(N + T - 1) / T, T, 0, stream>>>(hagg, W2, h2, N);
  // 6) out = b2; scatter-add messages
  k_init_b2<<<(2 * N + T - 1) / T, T, 0, stream>>>(out, b2, 2 * N);
  k_agg2<<<(E + N + T - 1) / T, T, 0, stream>>>(h2, src, dst, dinv, out, E, N);
}
